// lstm_encoder_70231305224529
// MI455X (gfx1250) — matrix operand order repaired and checked by reading, not compiled, not run on hardware
//
#include <hip/hip_runtime.h>
#include <hip/hip_bf16.h>

// ---------------- problem constants (fixed by the reference) ----------------
#define BSZ   4096
#define HSZ   512
#define LSZ   2
#define NSS   8
#define LATS  256
#define INDIM 30

typedef __attribute__((ext_vector_type(16))) __bf16 v16bf;
typedef __attribute__((ext_vector_type(2)))  __bf16 v2bf;
typedef __attribute__((ext_vector_type(8)))  float  v8f;

static __device__ __forceinline__ v8f wmma_bf16(v16bf a, v16bf b, v8f c) {
  // D = A(16x32 bf16) * B(32x16 bf16) + C(f32), wave32
  return __builtin_amdgcn_wmma_f32_16x16x32_bf16(
      /*neg_a=*/false, a, /*neg_b=*/false, b,
      /*c_mod=*/(short)0, c, /*reuse_a=*/false, /*reuse_b=*/false);
}

static __device__ __forceinline__ float sigmoidf_(float x) {
  return 1.0f / (1.0f + expf(-x));
}

// Split two f32 into packed bf16 hi dword + bf16 lo (residual) dword.
// The two converts pack into one dword; hi-as-f32 is a pure bit
// shift/mask of the packed dword (bf16 -> f32 == <<16).
static __device__ __forceinline__ void pack2_bf16_split(float a0, float a1,
                                                        unsigned& hpack,
                                                        unsigned& lpack) {
  v2bf hp;
  hp[0] = (__bf16)a0;
  hp[1] = (__bf16)a1;
  unsigned h32 = __builtin_bit_cast(unsigned, hp);
  float h0 = __builtin_bit_cast(float, h32 << 16);
  float h1 = __builtin_bit_cast(float, h32 & 0xFFFF0000u);
  v2bf lp;
  lp[0] = (__bf16)(a0 - h0);
  lp[1] = (__bf16)(a1 - h1);
  hpack = h32;
  lpack = __builtin_bit_cast(unsigned, lp);
}

// ============================================================================
// GEMM with bf16x3 split (fp32-class accuracy on bf16 WMMA).
//   C[M,N] = A[M,K] @ B[K,N] (+bias1+bias2) (+= existing C if accumulate)
//   transB=1 -> W is [N,K] row-major (PyTorch weight, B=W^T)
//   transB=0 -> W is [K,N] row-major
// REQUIRES: M % 64 == 0, N % 64 == 0, K even, grid exactly covers M x N.
// Tile 64x64, 256 threads = 8 waves; wave(wm,wn) owns rows wm*16..+16,
// cols wn*32..+32 (two 16x16 f32 accumulators).
// LDS tiles are FRAGMENT-MAJOR: [subtile16][lane][8 dwords], so each lane's
// v16bf operand is 32 contiguous bytes -> 2x ds_load_b128, no u16 gathers.
// Layout math (ISA 7.12.2, bf16):
//   A(m, k=2p):  lane = m + 16*bit2(p),  dword = 4*(p>>3) + (p&3)
//   B(k=2p, n):  lane = (n&15) + 16*bit2(p), dword = 4*(p>>3) + (p&3)
// Staging is two-phase (batch all global b64 loads, then convert+ds_store)
// so the loads overlap under one loadcnt wait.
// ============================================================================
#define KT 32
#define BM 64
#define BN 64

__global__ __launch_bounds__(256)
void wmma_gemm_bf16x3(const float* __restrict__ A, int lda,
                      const float* __restrict__ W, int ldb, int transB,
                      const float* __restrict__ bias1,
                      const float* __restrict__ bias2,
                      float* __restrict__ C, int ldc,
                      int M, int N, int K, int accumulate)
{
  __shared__ __align__(16) unsigned int sAh[4][32][8];  // 4 row-subtiles
  __shared__ __align__(16) unsigned int sAl[4][32][8];
  __shared__ __align__(16) unsigned int sBh[4][32][8];  // 4 col-subtiles
  __shared__ __align__(16) unsigned int sBl[4][32][8];

  const int tid  = threadIdx.x;
  const int wave = tid >> 5;
  const int lane = tid & 31;
  const int wm   = wave >> 1;   // 0..3 : 16-row subtile
  const int wn   = wave & 1;    // 0..1 : 32-col half (2 col-subtiles)

  const int row0 = blockIdx.y * BM;
  const int col0 = blockIdx.x * BN;

  v8f acc0 = {};
  v8f acc1 = {};

  const int nk = (K + KT - 1) / KT;
  for (int kt = 0; kt < nk; ++kt) {
    const int k0 = kt * KT;

    // ---- phase 1: batch all global loads for this K-tile ----
    float2 aval[4];
    bool   aok[4];
    #pragma unroll
    for (int it = 0; it < 4; ++it) {
      int i = tid + it * 256;
      int r = i >> 4;                    // 0..63
      int p = i & 15;                    // K-pair
      int gk = k0 + 2 * p;
      aok[it] = (gk < K);
      int gks = aok[it] ? gk : 0;        // clamp address, load unconditional
      aval[it] = *(const float2*)&A[(long)(row0 + r) * lda + gks];
    }
    float2 bval[4];
    bool   bok[4];
    if (transB) {
      #pragma unroll
      for (int it = 0; it < 4; ++it) {
        int i = tid + it * 256;
        int n = i >> 4;                  // coalesced along K of W[N,K]
        int p = i & 15;
        int gk = k0 + 2 * p;
        bok[it] = (gk < K);
        int gks = bok[it] ? gk : 0;
        bval[it] = *(const float2*)&W[(long)(col0 + n) * ldb + gks];
      }
    } else {
      #pragma unroll
      for (int it = 0; it < 4; ++it) {
        int i = tid + it * 256;
        int p = i >> 6;                  // 0..15
        int n = i & 63;                  // coalesced along N of W[K,N]
        int gk = k0 + 2 * p;
        bok[it] = (gk < K);
        long rb = (long)(bok[it] ? gk : 0) * ldb + (col0 + n);
        bval[it].x = W[rb];
        bval[it].y = W[rb + ldb];        // gk even, K even: gk+1 < K when ok
      }
    }
    // prefetch next A K-tile into cache (global_prefetch_b8)
    if (kt + 1 < nk) {
      int r = tid >> 2, c = (tid & 3) << 3;
      int gc = k0 + KT + c;
      if (gc < K) __builtin_prefetch(&A[(long)(row0 + r) * lda + gc], 0, 1);
    }

    // ---- phase 2: convert + fragment-major LDS stores ----
    #pragma unroll
    for (int it = 0; it < 4; ++it) {
      int i = tid + it * 256;
      int r = i >> 4;
      int p = i & 15;
      float a0 = aok[it] ? aval[it].x : 0.0f;
      float a1 = aok[it] ? aval[it].y : 0.0f;
      unsigned hp, lp;
      pack2_bf16_split(a0, a1, hp, lp);
      int sub = r >> 4;
      int ln  = (r & 15) + (((p >> 2) & 1) << 4);
      int v   = ((p >> 3) << 2) + (p & 3);
      sAh[sub][ln][v] = hp;
      sAl[sub][ln][v] = lp;
    }
    #pragma unroll
    for (int it = 0; it < 4; ++it) {
      int i = tid + it * 256;
      int n, p;
      if (transB) { n = i >> 4; p = i & 15; }
      else        { p = i >> 6; n = i & 63; }
      float b0 = bok[it] ? bval[it].x : 0.0f;
      float b1 = bok[it] ? bval[it].y : 0.0f;
      unsigned hp, lp;
      pack2_bf16_split(b0, b1, hp, lp);
      int sub = n >> 4;
      int ln  = (n & 15) + (((p >> 2) & 1) << 4);
      int v   = ((p >> 3) << 2) + (p & 3);
      sBh[sub][ln][v] = hp;
      sBl[sub][ln][v] = lp;
    }
    __syncthreads();

    // ---- fragment loads: 32 contiguous bytes per operand per lane ----
    v16bf ah  = *(const v16bf*)(&sAh[wm][lane][0]);
    v16bf al  = *(const v16bf*)(&sAl[wm][lane][0]);
    v16bf b0h = *(const v16bf*)(&sBh[wn * 2][lane][0]);
    v16bf b0l = *(const v16bf*)(&sBl[wn * 2][lane][0]);
    v16bf b1h = *(const v16bf*)(&sBh[wn * 2 + 1][lane][0]);
    v16bf b1l = *(const v16bf*)(&sBl[wn * 2 + 1][lane][0]);

    // bf16x3: hi*hi + hi*lo + lo*hi  (lo*lo is below the f32 result lsb)
    acc0 = wmma_bf16(ah, b0h, acc0);
    acc0 = wmma_bf16(ah, b0l, acc0);
    acc0 = wmma_bf16(al, b0h, acc0);
    acc1 = wmma_bf16(ah, b1h, acc1);
    acc1 = wmma_bf16(ah, b1l, acc1);
    acc1 = wmma_bf16(al, b1h, acc1);
    __syncthreads();
  }

  // ---- epilogue: bias, optional accumulate, store ----
  const int m     = lane & 15;
  const int hh    = lane >> 4;
  const int ccol0 = col0 + wn * 32 + m;
  const int ccol1 = ccol0 + 16;
  float bs0 = 0.0f, bs1 = 0.0f;
  if (bias1) { bs0 += bias1[ccol0]; bs1 += bias1[ccol1]; }
  if (bias2) { bs0 += bias2[ccol0]; bs1 += bias2[ccol1]; }
  #pragma unroll
  for (int r = 0; r < 8; ++r) {
    int rr = row0 + wm * 16 + r + (hh << 3);
    long i0 = (long)rr * ldc + ccol0;
    long i1 = (long)rr * ldc + ccol1;
    float v0 = acc0[r] + bs0;
    float v1 = acc1[r] + bs1;
    if (accumulate) { v0 += C[i0]; v1 += C[i1]; }
    C[i0] = v0;
    C[i1] = v1;
  }
}

// ============================================================================
// Feature extractor: x[b,0:32] = relu embeds (30 used, 2 zero pad)
// ============================================================================
__global__ __launch_bounds__(256)
void features_kernel(const float* __restrict__ obs, const float* __restrict__ act,
                     const float* __restrict__ rew,
                     const float* __restrict__ Wo, const float* __restrict__ bo,
                     const float* __restrict__ Wa, const float* __restrict__ ba,
                     const float* __restrict__ Wr, const float* __restrict__ br,
                     float* __restrict__ x)
{
  int b = blockIdx.x * blockDim.x + threadIdx.x;
  if (b >= BSZ) return;
  const float* o = obs + (long)b * 27;
  const float* a = act + (long)b * 8;
  float r0 = rew[b];
  float* xb = x + (long)b * 32;
  #pragma unroll
  for (int j = 0; j < 16; ++j) {
    float s = bo[j];
    for (int k = 0; k < 27; ++k) s += Wo[j * 27 + k] * o[k];
    xb[j] = fmaxf(s, 0.0f);
  }
  #pragma unroll
  for (int j = 0; j < 10; ++j) {
    float s = ba[j];
    for (int k = 0; k < 8; ++k) s += Wa[j * 8 + k] * a[k];
    xb[16 + j] = fmaxf(s, 0.0f);
  }
  #pragma unroll
  for (int j = 0; j < 4; ++j)
    xb[26 + j] = fmaxf(br[j] + Wr[j] * r0, 0.0f);
  xb[30] = 0.0f; xb[31] = 0.0f;
}

// ============================================================================
// Attention scores + softmax + argmax(context) -> pos[b].
// dot[b,n] = K_std[b,n,:] . Qk[b,:]   (bk.Qp term is constant over n: cancels
// in softmax and leaves V unchanged -> argmax invariant, so it is dropped)
// One wave per batch row; lanes stride the 2H=1024 reduction.
// ============================================================================
__global__ __launch_bounds__(256)
void attention_pos_kernel(const float* __restrict__ saved_h,
                          const float* __restrict__ saved_c,
                          const float* __restrict__ Qk,
                          const float* __restrict__ Wv,
                          const float* __restrict__ bv,
                          int* __restrict__ pos)
{
  int wave = threadIdx.x >> 5, lane = threadIdx.x & 31;
  int b = blockIdx.x * 8 + wave;
  const float* qk = Qk + (long)b * (2 * HSZ);
  float dots[NSS], vals[NSS];
  #pragma unroll
  for (int n = 0; n < NSS; ++n) {
    const float* ph = saved_h + ((((long)n * LSZ) + (LSZ - 1)) * BSZ + b) * HSZ;
    const float* pc = saved_c + ((((long)n * LSZ) + (LSZ - 1)) * BSZ + b) * HSZ;
    float d = 0.0f, vv = 0.0f;
    for (int g = lane; g < HSZ; g += 32) {
      float kh = ph[g];
      float kc = pc[g];
      d  += kh * qk[g] + kc * qk[HSZ + g];
      vv += kh * Wv[g] + kc * Wv[HSZ + g];
    }
    #pragma unroll
    for (int off = 16; off > 0; off >>= 1) {
      d  += __shfl_xor(d, off, 32);
      vv += __shfl_xor(vv, off, 32);
    }
    dots[n] = d;
    vals[n] = vv + bv[0];
  }
  if (lane == 0) {
    float mx = dots[0];
    #pragma unroll
    for (int n = 1; n < NSS; ++n) mx = fmaxf(mx, dots[n]);
    float den = 0.0f;
    #pragma unroll
    for (int n = 0; n < NSS; ++n) den += expf(dots[n] - mx);
    int best = 0; float bestv = -3.4e38f;
    #pragma unroll
    for (int n = 0; n < NSS; ++n) {
      float ctx = (expf(dots[n] - mx) / den) * vals[n];
      if (ctx > bestv) { bestv = ctx; best = n; }   // first-max like jnp.argmax
    }
    pos[b] = best;
  }
}

// ============================================================================
// Gather chosen (h,c) state: h0[l,b,:] = saved_h[pos[b], l, b, :]
// ============================================================================
__global__ __launch_bounds__(256)
void gather_kernel(const float* __restrict__ saved_h,
                   const float* __restrict__ saved_c,
                   const int* __restrict__ pos,
                   float* __restrict__ h0, float* __restrict__ c0)
{
  long t = (long)blockIdx.x * blockDim.x + threadIdx.x;
  if (t >= (long)LSZ * BSZ * HSZ) return;
  int g = (int)(t % HSZ);
  long r = t / HSZ;
  int b = (int)(r % BSZ);
  int l = (int)(r / BSZ);
  int p = pos[b];
  long src = ((((long)p * LSZ) + l) * BSZ + b) * HSZ + g;
  h0[t] = saved_h[src];
  c0[t] = saved_c[src];
}

// ============================================================================
// LSTM gating (PyTorch i,f,g,o order)
// ============================================================================
__global__ __launch_bounds__(256)
void lstm_cell_kernel(const float* __restrict__ gates,
                      const float* __restrict__ c_prev,
                      float* __restrict__ h_out, float* __restrict__ c_out)
{
  long t = (long)blockIdx.x * blockDim.x + threadIdx.x;
  if (t >= (long)BSZ * HSZ) return;
  int b = (int)(t / HSZ), j = (int)(t % HSZ);
  const float* g = gates + (long)b * 4 * HSZ;
  float ig = sigmoidf_(g[j]);
  float fg = sigmoidf_(g[HSZ + j]);
  float gg = tanhf(g[2 * HSZ + j]);
  float og = sigmoidf_(g[3 * HSZ + j]);
  float c2 = fg * c_prev[t] + ig * gg;
  c_out[t] = c2;
  h_out[t] = og * tanhf(c2);
}

// ============================================================================
// Reparameterization: sample = eps * exp(0.5*logvar) + mu
// ============================================================================
__global__ __launch_bounds__(256)
void sample_kernel(const float* __restrict__ mu, const float* __restrict__ lv,
                   const float* __restrict__ eps, float* __restrict__ out)
{
  long t = (long)blockIdx.x * blockDim.x + threadIdx.x;
  if (t >= (long)BSZ * LATS) return;
  out[t] = eps[t] * expf(0.5f * lv[t]) + mu[t];
}

// ============================================================================
extern "C" void kernel_launch(void* const* d_in, const int* in_sizes, int n_in,
                              void* d_out, int out_size, void* d_ws, size_t ws_size,
                              hipStream_t stream)
{
  const float* obs     = (const float*)d_in[0];
  const float* action  = (const float*)d_in[1];
  const float* reward  = (const float*)d_in[2];
  const float* saved_h = (const float*)d_in[3];
  const float* saved_c = (const float*)d_in[4];
  const float* W_obs   = (const float*)d_in[5];
  const float* b_obs   = (const float*)d_in[6];
  const float* W_act   = (const float*)d_in[7];
  const float* b_act   = (const float*)d_in[8];
  const float* W_rew   = (const float*)d_in[9];
  const float* b_rew   = (const float*)d_in[10];
  const float* Wq      = (const float*)d_in[11];
  const float* bq      = (const float*)d_in[12];
  const float* Wk      = (const float*)d_in[13];
  // d_in[14] = bk : provably irrelevant to outputs (constant softmax shift)
  const float* Wv      = (const float*)d_in[15];
  const float* bv      = (const float*)d_in[16];
  const float* W_ih0   = (const float*)d_in[17];
  const float* W_hh0   = (const float*)d_in[18];
  const float* b_ih0   = (const float*)d_in[19];
  const float* b_hh0   = (const float*)d_in[20];
  const float* W_ih1   = (const float*)d_in[21];
  const float* W_hh1   = (const float*)d_in[22];
  const float* b_ih1   = (const float*)d_in[23];
  const float* b_hh1   = (const float*)d_in[24];
  const float* W_mu    = (const float*)d_in[25];
  const float* b_mu    = (const float*)d_in[26];
  const float* W_lv    = (const float*)d_in[27];
  const float* b_lv    = (const float*)d_in[28];
  const float* eps     = (const float*)d_in[29];

  // workspace layout (floats)
  float* ws = (float*)d_ws;
  const long nx  = (long)BSZ * 32;              // padded features
  const long nqp = (long)BSZ * 2 * HSZ;         // Qp [B,1024]
  float* x     = ws;
  float* Qp    = x + nx;
  float* Qkbuf = Qp + nqp;
  float* gates = Qp;                            // [B,4H] aliases Qp+Qk (dead after attention)
  float* h0    = Qkbuf + nqp;                   // [L,B,H]
  float* c0    = h0 + (long)LSZ * BSZ * HSZ;
  int*   pos   = (int*)(c0 + (long)LSZ * BSZ * HSZ);

  // output layout (floats): mu | logvar | sample | h_fin[2,B,H] | c_fin[2,B,H]
  float* out   = (float*)d_out;
  float* o_mu  = out;
  float* o_lv  = o_mu + (long)BSZ * LATS;
  float* o_sm  = o_lv + (long)BSZ * LATS;
  float* o_h1  = o_sm + (long)BSZ * LATS;
  float* o_h2  = o_h1 + (long)BSZ * HSZ;
  float* o_c1  = o_h2 + (long)BSZ * HSZ;
  float* o_c2  = o_c1 + (long)BSZ * HSZ;

  // 1) features -> x [B,32] (30 used)
  features_kernel<<<BSZ / 256, 256, 0, stream>>>(obs, action, reward,
      W_obs, b_obs, W_act, b_act, W_rew, b_rew, x);

  // 2) Qp = x @ Wq^T + bq            [B,1024]
  wmma_gemm_bf16x3<<<dim3(1024 / BN, BSZ / BM), 256, 0, stream>>>(
      x, 32, Wq, INDIM, 1, bq, nullptr, Qp, 1024, BSZ, 1024, INDIM, 0);

  // 3) Qk = Qp @ Wk                  [B,1024]  (folds Wk into query: 8x FLOP cut)
  wmma_gemm_bf16x3<<<dim3(1024 / BN, BSZ / BM), 256, 0, stream>>>(
      Qp, 1024, Wk, 1024, 0, nullptr, nullptr, Qkbuf, 1024, BSZ, 1024, 1024, 0);

  // 4) attention scores, softmax, argmax -> pos[b]
  attention_pos_kernel<<<BSZ / 8, 256, 0, stream>>>(saved_h, saved_c, Qkbuf, Wv, bv, pos);

  // 5) gather chosen states
  gather_kernel<<<(LSZ * BSZ * HSZ) / 256, 256, 0, stream>>>(saved_h, saved_c, pos, h0, c0);

  // 6) LSTM layer 0: gates = x@W_ih0^T + h0[0]@W_hh0^T + b_ih0 + b_hh0
  wmma_gemm_bf16x3<<<dim3(2048 / BN, BSZ / BM), 256, 0, stream>>>(
      x, 32, W_ih0, INDIM, 1, b_ih0, b_hh0, gates, 2048, BSZ, 2048, INDIM, 0);
  wmma_gemm_bf16x3<<<dim3(2048 / BN, BSZ / BM), 256, 0, stream>>>(
      h0, HSZ, W_hh0, HSZ, 1, nullptr, nullptr, gates, 2048, BSZ, 2048, HSZ, 1);
  lstm_cell_kernel<<<(BSZ * HSZ) / 256, 256, 0, stream>>>(gates, c0, o_h1, o_c1);

  // 7) LSTM layer 1
  wmma_gemm_bf16x3<<<dim3(2048 / BN, BSZ / BM), 256, 0, stream>>>(
      o_h1, HSZ, W_ih1, HSZ, 1, b_ih1, b_hh1, gates, 2048, BSZ, 2048, HSZ, 0);
  wmma_gemm_bf16x3<<<dim3(2048 / BN, BSZ / BM), 256, 0, stream>>>(
      h0 + (long)BSZ * HSZ, HSZ, W_hh1, HSZ, 1, nullptr, nullptr, gates, 2048, BSZ, 2048, HSZ, 1);
  lstm_cell_kernel<<<(BSZ * HSZ) / 256, 256, 0, stream>>>(
      gates, c0 + (long)BSZ * HSZ, o_h2, o_c2);

  // 8) VAE head
  wmma_gemm_bf16x3<<<dim3(LATS / BN, BSZ / BM), 256, 0, stream>>>(
      o_h2, HSZ, W_mu, HSZ, 1, b_mu, nullptr, o_mu, LATS, BSZ, LATS, HSZ, 0);
  wmma_gemm_bf16x3<<<dim3(LATS / BN, BSZ / BM), 256, 0, stream>>>(
      o_h2, HSZ, W_lv, HSZ, 1, b_lv, nullptr, o_lv, LATS, BSZ, LATS, HSZ, 0);

  // 9) reparameterize
  sample_kernel<<<(BSZ * LATS) / 256, 256, 0, stream>>>(o_mu, o_lv, eps, o_sm);
}
